// BiaffineSpanHead_56848187129953
// MI455X (gfx1250) — matrix operand order repaired and checked by reading, not compiled, not run on hardware
//
#include <hip/hip_runtime.h>
#include <hip/hip_bf16.h>

// ---------------------------------------------------------------------------
// BiaffineSpanHead on MI455X (gfx1250): bf16 WMMA pipeline, fp32 accumulate.
// B=4, S=1024, IN=1024, H=256, C=8. ~26 GFLOP of GEMM vs 134 MB of output
// => compute-bound => v_wmma_f32_16x16x32_bf16 everywhere.
// Round 2: biaff A-panel staged in LDS (XOR-swizzled, conflict-minimal),
// shared by all 8 waves of a block -> 8x less L2 read traffic.
// ---------------------------------------------------------------------------

typedef __attribute__((ext_vector_type(16))) __bf16 v16bf;
typedef __attribute__((ext_vector_type(8)))  float  v8f;
typedef __attribute__((ext_vector_type(4)))  unsigned int u32x4;
typedef __attribute__((ext_vector_type(4)))  float  f32x4;

struct B128x2 { u32x4 lo, hi; };

static __device__ __forceinline__ __bf16 f2bf(float f) {
    unsigned u = __builtin_bit_cast(unsigned, f);
    unsigned r = u + 0x7FFFu + ((u >> 16) & 1u);   // round-to-nearest-even
    unsigned short h = (unsigned short)(r >> 16);
    return __builtin_bit_cast(__bf16, h);
}
static __device__ __forceinline__ float bf2f(__bf16 b) {
    unsigned short h = __builtin_bit_cast(unsigned short, b);
    unsigned u = ((unsigned)h) << 16;
    return __builtin_bit_cast(float, u);
}

// A operand: 16x32 bf16 tile, row-major source, row stride lda (elements).
// Lane L: row = L&15, K = {h8..h8+7, 16+h8..+7} with h8 = (L>>4)*8.
static __device__ __forceinline__ v16bf load_frag_A(const __bf16* __restrict__ tile,
                                                    int lda, int lane) {
    int half = lane >> 4, r = lane & 15;
    const __bf16* p = tile + (size_t)r * lda + half * 8;
    B128x2 t;
    t.lo = *(const u32x4*)(p);        // K = half*8 .. +7      (16 B)
    t.hi = *(const u32x4*)(p + 16);   // K = 16+half*8 .. +7   (16 B)
    return __builtin_bit_cast(v16bf, t);
}

// B operand: 32x16 bf16 (KxN). Column n of B == row n of the row-major source.
// Lane L: col n = L&15, K = {h8..h8+7, 16+h8..+7} with h8 = (L>>4)*8.
static __device__ __forceinline__ v16bf load_frag_B(const __bf16* __restrict__ tile,
                                                    int ldb, int lane) {
    int half = lane >> 4, n = lane & 15;
    const __bf16* p = tile + (size_t)n * ldb + half * 8;
    B128x2 t;
    t.lo = *(const u32x4*)(p);
    t.hi = *(const u32x4*)(p + 16);
    return __builtin_bit_cast(v16bf, t);
}

static __device__ __forceinline__ v8f wmma_bf16(v16bf a, v16bf b, v8f c) {
    return __builtin_amdgcn_wmma_f32_16x16x32_bf16(false, a, false, b,
                                                   (short)0, c, false, false);
}

// LDS A-panel addressing: logical (c, row r, 16B-unit x in 0..31) -> physical
// 16B unit. XOR-swizzle low nibble of x with r: the 16 lanes of a half-wave
// hit 16 distinct 4-bank groups (2-cycle minimum for a 512B wave read).
static __device__ __forceinline__ int lds_unit(int c, int r, int x) {
    int xs = (x & 16) | ((x ^ r) & 15);
    return (c << 9) | (r << 5) | xs;          // units of 16 bytes, 4096 total
}

// ------------------------------- converters --------------------------------

__global__ void cvt_f32_bf16_kernel(const float* __restrict__ src,
                                    __bf16* __restrict__ dst, int n) {
    int i = blockIdx.x * blockDim.x + threadIdx.x;
    if (i < n) dst[i] = f2bf(src[i]);
}

// U (h, c, g) f32  ->  Ut (c, g, h) bf16  (so B-columns g are contiguous rows)
__global__ void permU_kernel(const float* __restrict__ U, __bf16* __restrict__ Ut) {
    int i = blockIdx.x * blockDim.x + threadIdx.x;
    if (i >= 8 * 256 * 256) return;
    int h = i & 255, g = (i >> 8) & 255, c = i >> 16;
    Ut[i] = f2bf(U[(size_t)h * 2048 + c * 256 + g]);
}

// ------------------------- projection GEMM (K=1024) ------------------------
// out[m, n] = sum_k seq[m,k] * w[n,k] + bias[n]    M=4096 N=256 K=1024
__global__ void proj_kernel(const __bf16* __restrict__ seqb,
                            const __bf16* __restrict__ wb,
                            const float*  __restrict__ bias,
                            __bf16* __restrict__ outb) {
    int lane = threadIdx.x & 31;
    int wave = threadIdx.x >> 5;
    int w = blockIdx.x * 8 + wave;
    int m0 = (w >> 4) * 16;
    int n0 = (w & 15) * 16;
    v8f acc = {};
    for (int k0 = 0; k0 < 1024; k0 += 32) {
        v16bf a = load_frag_A(seqb + (size_t)m0 * 1024 + k0, 1024, lane);
        v16bf b = load_frag_B(wb   + (size_t)n0 * 1024 + k0, 1024, lane);
        acc = wmma_bf16(a, b, acc);
    }
    int N = lane & 15, half = lane >> 4;
    float bv = bias[n0 + N];
#pragma unroll
    for (int i = 0; i < 8; ++i) {
        int m = m0 + i + 8 * half;
        outb[(size_t)m * 256 + n0 + N] = f2bf(acc[i] + bv);
    }
}

// ------------------------ V[b,c] = Hs[b] @ U_c  (K=256) --------------------
__global__ void vproj_kernel(const __bf16* __restrict__ Hsb,   // (B,1024,256)
                             const __bf16* __restrict__ Utb,   // (C,256,256)
                             __bf16* __restrict__ Vb) {        // (B,C,1024,256)
    int lane = threadIdx.x & 31;
    int wave = threadIdx.x >> 5;
    int c = blockIdx.y, b = blockIdx.z;
    int w = blockIdx.x * 8 + wave;            // 0..1023
    int m0 = (w >> 4) * 16;
    int n0 = (w & 15) * 16;
    const __bf16* A  = Hsb + (size_t)b * 1024 * 256;
    const __bf16* Bm = Utb + (size_t)c * 256 * 256;
    v8f acc = {};
    for (int k0 = 0; k0 < 256; k0 += 32) {
        v16bf a  = load_frag_A(A  + (size_t)m0 * 256 + k0, 256, lane);
        v16bf bf = load_frag_B(Bm + (size_t)n0 * 256 + k0, 256, lane);
        acc = wmma_bf16(a, bf, acc);
    }
    __bf16* O = Vb + ((size_t)b * 8 + c) * 1024 * 256;
    int N = lane & 15, half = lane >> 4;
#pragma unroll
    for (int i = 0; i < 8; ++i) {
        int m = m0 + i + 8 * half;
        O[(size_t)m * 256 + n0 + N] = f2bf(acc[i]);
    }
}

// --------------------------- tiny linear terms -----------------------------
// lin_s[row,c] = Hs[row,:].Ws[c,:]   lin_e[row,c] = He[row,:].We[c,:] + Wb[c]
__global__ void lin_kernel(const __bf16* __restrict__ Hsb,
                           const __bf16* __restrict__ Heb,
                           const float* __restrict__ W,      // (8, 512)
                           const float* __restrict__ Wbias,  // (8)
                           float* __restrict__ lin_s,
                           float* __restrict__ lin_e) {
    int t = blockIdx.x * blockDim.x + threadIdx.x;
    if (t >= 4 * 1024 * 8) return;
    int c = t & 7;
    int row = t >> 3;
    const float* Wr = W + c * 512;
    const __bf16* hs = Hsb + (size_t)row * 256;
    const __bf16* he = Heb + (size_t)row * 256;
    float ss = 0.f, se = 0.f;
    for (int h = 0; h < 256; ++h) {
        ss += bf2f(hs[h]) * Wr[h];
        se += bf2f(he[h]) * Wr[256 + h];
    }
    lin_s[t] = ss;
    lin_e[t] = se + Wbias[c];
}

// ------------------- out[b,s,e,c] = V[b,c,s,:].He[b,e,:] -------------------
// + lin_s[b,s,c] + lin_e[b,e,c]   (W_bias folded into lin_e).
// Block = 8 waves, one 16-row s-tile, 128 e columns, all 8 c.
// A panel (8c x 16rows x K=256 = 64 KB) staged once in LDS, shared by all
// waves; WMMA A-fragments come from ds_load_b128 (swizzled, conflict-minimal).
__global__ void biaff_kernel(const __bf16* __restrict__ Vb,    // (B,8,1024,256)
                             const __bf16* __restrict__ Heb,   // (B,1024,256)
                             const float* __restrict__ lin_s,  // (B,1024,8)
                             const float* __restrict__ lin_e,  // (B,1024,8)
                             float* __restrict__ out) {        // (B,1024,1024,8)
    extern __shared__ char smem[];                 // 65536 bytes
    u32x4* lds = (u32x4*)smem;

    int tid  = threadIdx.x;
    int lane = tid & 31;
    int wave = tid >> 5;
    int b  = blockIdx.z;
    int s0 = blockIdx.x * 16;
    int e0 = (blockIdx.y * 8 + wave) * 16;

    // ---- cooperative panel load: V[b, 0..7, s0..s0+15, :] -> LDS ----------
    // 4096 16-byte units; rows s0..s0+15 of each V[b,c] are contiguous 8 KB.
    for (int u = tid; u < 4096; u += 256) {
        int c = u >> 9;
        int r = (u >> 5) & 15;
        int x = u & 31;
        const u32x4* src =
            (const u32x4*)(Vb + (((size_t)b * 8 + c) * 1024 + s0 + r) * 256 + x * 8);
        lds[lds_unit(c, r, x)] = *src;
    }
    __syncthreads();

    const __bf16* Bsrc = Heb + (size_t)b * 1024 * 256 + (size_t)e0 * 256;
    int half = lane >> 4, r = lane & 15;

    v8f acc[8] = {};
    for (int k0 = 0; k0 < 256; k0 += 32) {
        v16bf bf = load_frag_B(Bsrc + k0, 256, lane);
        int x0 = (k0 >> 3) + half;                 // lo 16B unit: K=k0+half*8..
        int x1 = x0 + 2;                           // hi 16B unit: K=k0+16+half*8..
#pragma unroll
        for (int c = 0; c < 8; ++c) {
            B128x2 t;
            t.lo = lds[lds_unit(c, r, x0)];
            t.hi = lds[lds_unit(c, r, x1)];
            v16bf a = __builtin_bit_cast(v16bf, t);
            acc[c] = wmma_bf16(a, bf, acc[c]);
        }
    }

    int N = lane & 15;
    const f32x4* le4 = (const f32x4*)(lin_e + ((size_t)b * 1024 + e0 + N) * 8);
    f32x4 le0 = le4[0], le1 = le4[1];
#pragma unroll
    for (int i = 0; i < 8; ++i) {
        int s = s0 + i + 8 * half;
        const f32x4* ls4 = (const f32x4*)(lin_s + ((size_t)b * 1024 + s) * 8);
        f32x4 ls0 = ls4[0], ls1 = ls4[1];
        f32x4 o0, o1;
        o0.x = acc[0][i] + ls0.x + le0.x;
        o0.y = acc[1][i] + ls0.y + le0.y;
        o0.z = acc[2][i] + ls0.z + le0.z;
        o0.w = acc[3][i] + ls0.w + le0.w;
        o1.x = acc[4][i] + ls1.x + le1.x;
        o1.y = acc[5][i] + ls1.y + le1.y;
        o1.z = acc[6][i] + ls1.z + le1.z;
        o1.w = acc[7][i] + ls1.w + le1.w;
        f32x4* op = (f32x4*)(out + (((size_t)b * 1024 + s) * 1024 + e0 + N) * 8);
        op[0] = o0;   // c = 0..3  -> contiguous 32 B
        op[1] = o1;   // c = 4..7
    }
}

// ---------------------------------------------------------------------------

extern "C" void kernel_launch(void* const* d_in, const int* in_sizes, int n_in,
                              void* d_out, int out_size, void* d_ws, size_t ws_size,
                              hipStream_t stream) {
    (void)in_sizes; (void)n_in; (void)out_size; (void)ws_size;
    const float* seq = (const float*)d_in[0];   // (4,1024,1024)
    const float* U   = (const float*)d_in[1];   // (256,8,256)
    const float* W   = (const float*)d_in[2];   // (8,512)
    const float* Wb  = (const float*)d_in[3];   // (8)
    const float* sw  = (const float*)d_in[4];   // (256,1024)
    const float* sb  = (const float*)d_in[5];   // (256)
    const float* ew  = (const float*)d_in[6];   // (256,1024)
    const float* eb  = (const float*)d_in[7];   // (256)
    float* out = (float*)d_out;

    char* ws = (char*)d_ws;
    size_t off = 0;
    auto take = [&](size_t bytes) -> char* {
        char* p = ws + off;
        off += (bytes + 255) & ~(size_t)255;
        return p;
    };
    __bf16* seqb = (__bf16*)take((size_t)4 * 1024 * 1024 * 2);      // 8 MB
    __bf16* swb  = (__bf16*)take((size_t)256 * 1024 * 2);
    __bf16* ewb  = (__bf16*)take((size_t)256 * 1024 * 2);
    __bf16* Utb  = (__bf16*)take((size_t)8 * 256 * 256 * 2);
    __bf16* Hsb  = (__bf16*)take((size_t)4 * 1024 * 256 * 2);
    __bf16* Heb  = (__bf16*)take((size_t)4 * 1024 * 256 * 2);
    __bf16* Vb   = (__bf16*)take((size_t)4 * 8 * 1024 * 256 * 2);   // 16.8 MB
    float*  lins = (float*)take((size_t)4 * 1024 * 8 * 4);
    float*  line = (float*)take((size_t)4 * 1024 * 8 * 4);

    // 1) precision conversion / layout
    cvt_f32_bf16_kernel<<<16384, 256, 0, stream>>>(seq, seqb, 4 * 1024 * 1024);
    cvt_f32_bf16_kernel<<<1024,  256, 0, stream>>>(sw, swb, 256 * 1024);
    cvt_f32_bf16_kernel<<<1024,  256, 0, stream>>>(ew, ewb, 256 * 1024);
    permU_kernel<<<2048, 256, 0, stream>>>(U, Utb);

    // 2) Hs / He projections (WMMA, K=1024)
    proj_kernel<<<512, 256, 0, stream>>>(seqb, swb, sb, Hsb);
    proj_kernel<<<512, 256, 0, stream>>>(seqb, ewb, eb, Heb);

    // 3) V[b,c] = Hs[b] @ U_c (WMMA, K=256)
    vproj_kernel<<<dim3(128, 8, 4), 256, 0, stream>>>(Hsb, Utb, Vb);

    // 4) linear terms (+bias)
    lin_kernel<<<128, 256, 0, stream>>>(Hsb, Heb, W, Wb, lins, line);

    // 5) big batched GEMM (17 GF) + fused epilogue:
    //    LDS-staged A panel (64 KB), coalesced b128 stores
    biaff_kernel<<<dim3(64, 8, 4), 256, 65536, stream>>>(Vb, Heb, lins, line, out);
}
